// LearnableRetrievalBranch_17746804867781
// MI455X (gfx1250) — matrix operand order repaired and checked by reading, not compiled, not run on hardware
//
#include <hip/hip_runtime.h>
#include <hip/hip_bf16.h>

// ---------------------------------------------------------------------------
// Problem constants (from the reference)
// ---------------------------------------------------------------------------
#define BATCH   512
#define NBANK   100000
#define DIM     256
#define TOPK    16

#define CHUNK_N   512                    // columns of N per block in main kernel
#define NCHUNK    196                    // ceil(100000 / 512)
#define CAND_PER_ROW (NCHUNK * TOPK)     // 3136 candidates per row

typedef __attribute__((ext_vector_type(16))) __bf16 v16bf;
typedef __attribute__((ext_vector_type(8)))  __bf16 v8bf;
typedef __attribute__((ext_vector_type(8)))  float  v8f;

#define NEG_INF (-__builtin_inff())

// Register-resident sorted-descending top-16 insertion (fully unrolled so the
// arrays stay in VGPRs).
#define TOP16_INSERT(ls, lc, sval, cval)                                       \
  do {                                                                         \
    float _s = (sval);                                                         \
    int   _c = (cval);                                                         \
    _Pragma("unroll")                                                          \
    for (int _j = 0; _j < 16; ++_j) {                                          \
      if (_s > (ls)[_j]) {                                                     \
        float _ts = (ls)[_j]; int _tc = (lc)[_j];                              \
        (ls)[_j] = _s; (lc)[_j] = _c;                                          \
        _s = _ts; _c = _tc;                                                    \
      }                                                                        \
    }                                                                          \
  } while (0)

#define TOP16_INIT(ls, lc)                                                     \
  do {                                                                         \
    _Pragma("unroll")                                                          \
    for (int _j = 0; _j < 16; ++_j) { (ls)[_j] = NEG_INF; (lc)[_j] = -1; }     \
  } while (0)

// ---------------------------------------------------------------------------
// Kernel 1: q = z_ego @ W^T, row L2-normalize, emit bf16.  grid=(512), blk=256
// ---------------------------------------------------------------------------
__global__ void __launch_bounds__(256)
lrb_proj_norm_q(const float* __restrict__ z,
                const float* __restrict__ W,
                __bf16* __restrict__ qn) {
  __shared__ float zrow[DIM];
  __shared__ float red[256];
  const int b = blockIdx.x;
  const int t = threadIdx.x;

  zrow[t] = z[(size_t)b * DIM + t];
  __syncthreads();

  // q[b, t] = sum_d z[b, d] * W[t, d]
  float acc = 0.f;
  #pragma unroll 8
  for (int d = 0; d < DIM; ++d)
    acc = fmaf(zrow[d], W[(size_t)t * DIM + d], acc);

  red[t] = acc * acc;
  __syncthreads();
  #pragma unroll
  for (int s = 128; s > 0; s >>= 1) {
    if (t < s) red[t] += red[t + s];
    __syncthreads();
  }
  const float inv = 1.0f / fmaxf(sqrtf(red[0]), 1e-12f);
  qn[(size_t)b * DIM + t] = (__bf16)(acc * inv);
}

// ---------------------------------------------------------------------------
// Kernel 2: row L2-normalize exo_bank, emit bf16.  grid=(100000), blk=256
// ---------------------------------------------------------------------------
__global__ void __launch_bounds__(256)
lrb_norm_bank(const float* __restrict__ bank,
              __bf16* __restrict__ bankn) {
  __shared__ float red[256];
  const int r = blockIdx.x;
  const int t = threadIdx.x;
  const float x = bank[(size_t)r * DIM + t];

  red[t] = x * x;
  __syncthreads();
  #pragma unroll
  for (int s = 128; s > 0; s >>= 1) {
    if (t < s) red[t] += red[t + s];
    __syncthreads();
  }
  const float inv = 1.0f / fmaxf(sqrtf(red[0]), 1e-12f);
  bankn[(size_t)r * DIM + t] = (__bf16)(x * inv);
}

// ---------------------------------------------------------------------------
// Kernel 3: fused WMMA GEMM (qn @ bankn^T), + 0.5*sim_vt, per-chunk top-16.
// grid = (NCHUNK, BATCH/16), block = 256 (8 waves). Each wave owns 4 16x16
// output tiles; K=256 accumulated with 8 x v_wmma_f32_16x16x32_bf16.
//
// WMMA is executed UNCONDITIONALLY (EXEC all-1s, as the ISA requires): for the
// few out-of-range column tiles in the last chunk we clamp the B row index to
// row 0 (per-lane v_cndmask, no EXEC change) and discard the garbage scores at
// the guarded LDS-store step.
// ---------------------------------------------------------------------------
__global__ void __launch_bounds__(256)
lrb_gemm_score_topk(const __bf16* __restrict__ qn,
                    const __bf16* __restrict__ bankn,
                    const float*  __restrict__ simvt,
                    float* __restrict__ candS,
                    int*   __restrict__ candI) {
  __shared__ float          sTile[16][CHUNK_N];   // 32 KB
  __shared__ float          cSc[16][256];         // 16 KB
  __shared__ unsigned short cCl[16][256];         //  8 KB

  const int tid       = threadIdx.x;
  // force the wave id into an SGPR so tile math is provably scalar
  const int wave      = __builtin_amdgcn_readfirstlane(tid) >> 5;  // 0..7
  const unsigned lane = tid & 31u;         // wave32
  const unsigned ln16 = lane & 15u;
  const unsigned half = lane >> 4;         // 0 or 1
  const int chunk     = blockIdx.x;
  const int chunkBase = chunk * CHUNK_N;
  const int rowBase   = blockIdx.y * 16;

  // init score tile to -inf (covers column tiles beyond N in last chunk)
  #pragma unroll
  for (int i = tid; i < 16 * CHUNK_N; i += 256)
    (&sTile[0][0])[i] = NEG_INF;
  __syncthreads();

  // ---- WMMA phase: 4 tiles per wave, tiles (wave*4 .. wave*4+3) ----
  v8f acc[4] = {};
  const int tile0 = wave * 4;

  // Per-tile B base pointers, with OOB bank rows clamped to row 0 (branchless).
  const __bf16* bBase[4];
  #pragma unroll
  for (int t = 0; t < 4; ++t) {
    int r = chunkBase + (tile0 + t) * 16 + (int)ln16;  // bank row for this lane
    r = (r < NBANK) ? r : 0;                           // clamp, no EXEC change
    bBase[t] = bankn + (size_t)r * DIM + half * 16;
  }
  const __bf16* aBase = qn + (size_t)(rowBase + ln16) * DIM + half * 16;

  #pragma unroll
  for (int kk = 0; kk < DIM; kk += 32) {
    // A operand: 16x32 bf16.  lane<16: K kk+0..15 of row ln16;
    // lane>=16: K kk+16..31  (contiguous, matches B).
    v8bf alo = *reinterpret_cast<const v8bf*>(aBase + kk);
    v8bf ahi = *reinterpret_cast<const v8bf*>(aBase + kk + 8);
    v16bf a;
    #pragma unroll
    for (int i = 0; i < 8; ++i) { a[i] = alo[i]; a[i + 8] = ahi[i]; }

    #pragma unroll
    for (int t = 0; t < 4; ++t) {
      // B operand: 32x16 bf16; lane<16 holds K kk+0..15, lane>=16 K kk+16..31.
      v16bf bm = *reinterpret_cast<const v16bf*>(bBase[t] + kk);
      acc[t] = __builtin_amdgcn_wmma_f32_16x16x32_bf16(
          false, a, false, bm, (short)0, acc[t], false, false);
    }
  }

  // ---- combine with sim_vt, write to LDS score tile (per-lane guarded) ----
  #pragma unroll
  for (int t = 0; t < 4; ++t) {
    const int colLoc = (tile0 + t) * 16 + (int)ln16;
    const int colG   = chunkBase + colLoc;
    if (colG < NBANK) {
      #pragma unroll
      for (int r = 0; r < 8; ++r) {
        const int m = (int)(half << 3) + r;   // C layout: M = r (+8 for hi half)
        const float sv = simvt[(size_t)(rowBase + m) * NBANK + colG];
        sTile[m][colLoc] = 0.5f * acc[t][r] + 0.5f * sv;
      }
    }
  }
  __syncthreads();

  // ---- per-slice local top-16: thread = (row, slice) ----
  {
    const int row = tid >> 4;
    const int sl  = tid & 15;
    float ls[16]; int lc[16];
    TOP16_INIT(ls, lc);
    for (int c = sl; c < CHUNK_N; c += 16)
      TOP16_INSERT(ls, lc, sTile[row][c], c);
    #pragma unroll
    for (int j = 0; j < 16; ++j) {
      cSc[row][sl * 16 + j] = ls[j];
      cCl[row][sl * 16 + j] = (unsigned short)(lc[j] < 0 ? 0xFFFF : lc[j]);
    }
  }
  __syncthreads();

  // ---- merge 256 -> 16 per row, write chunk candidates ----
  if (tid < 16) {
    const int row = tid;
    float ls[16]; int lc[16];
    TOP16_INIT(ls, lc);
    for (int i = 0; i < 256; ++i) {
      const int c = (cCl[row][i] == 0xFFFF) ? -1 : (int)cCl[row][i];
      TOP16_INSERT(ls, lc, cSc[row][i], c);
    }
    const size_t base =
        (size_t)(rowBase + row) * CAND_PER_ROW + (size_t)chunk * TOPK;
    #pragma unroll
    for (int j = 0; j < 16; ++j) {
      candS[base + j] = ls[j];
      candI[base + j] = (lc[j] < 0) ? -1 : (chunkBase + lc[j]);
    }
  }
}

// ---------------------------------------------------------------------------
// Kernel 4: per-row global merge (3136 -> 16, sorted desc) + gather + indices.
// grid = (512), block = 256.
// ---------------------------------------------------------------------------
__global__ void __launch_bounds__(256)
lrb_finalize(const float* __restrict__ candS,
             const int*   __restrict__ candI,
             const float* __restrict__ bank,
             float* __restrict__ outGather,
             float* __restrict__ outIdx) {
  __shared__ float mS[32 * 16];
  __shared__ int   mI[32 * 16];
  __shared__ int   topIdx[TOPK];

  const int b   = blockIdx.x;
  const int tid = threadIdx.x;
  const size_t cbase = (size_t)b * CAND_PER_ROW;

  if (tid < 32) {
    float ls[16]; int lc[16];
    TOP16_INIT(ls, lc);
    for (int i = tid; i < CAND_PER_ROW; i += 32)
      TOP16_INSERT(ls, lc, candS[cbase + i], candI[cbase + i]);
    #pragma unroll
    for (int j = 0; j < 16; ++j) { mS[tid * 16 + j] = ls[j]; mI[tid * 16 + j] = lc[j]; }
  }
  __syncthreads();

  if (tid == 0) {
    float ls[16]; int lc[16];
    TOP16_INIT(ls, lc);
    for (int i = 0; i < 32 * 16; ++i)
      TOP16_INSERT(ls, lc, mS[i], mI[i]);
    #pragma unroll
    for (int j = 0; j < TOPK; ++j) {
      int idx = lc[j];
      if (idx < 0) idx = 0;
      if (idx >= NBANK) idx = NBANK - 1;
      topIdx[j] = idx;
      outIdx[(size_t)b * TOPK + j] = (float)idx;   // indices emitted as floats
    }
  }
  __syncthreads();

  // gather ORIGINAL (un-normalized) bank rows
  #pragma unroll
  for (int j = 0; j < TOPK; ++j) {
    const int idx = topIdx[j];
    outGather[((size_t)b * TOPK + j) * DIM + tid] =
        bank[(size_t)idx * DIM + tid];
  }
}

// ---------------------------------------------------------------------------
// Host-side launcher
// ---------------------------------------------------------------------------
extern "C" void kernel_launch(void* const* d_in, const int* in_sizes, int n_in,
                              void* d_out, int out_size, void* d_ws, size_t ws_size,
                              hipStream_t stream) {
  const float* z_ego = (const float*)d_in[0];   // [512, 256]
  const float* bank  = (const float*)d_in[1];   // [100000, 256]
  const float* simvt = (const float*)d_in[2];   // [512, 100000]
  const float* W     = (const float*)d_in[3];   // [256, 256]
  // d_in[4] = k (always 16 after clamping to MAX_K)

  // workspace layout
  char* ws = (char*)d_ws;
  size_t off = 0;
  __bf16* qn    = (__bf16*)(ws + off); off += (size_t)BATCH * DIM * 2;      // 256 KB
  __bf16* bankn = (__bf16*)(ws + off); off += (size_t)NBANK * DIM * 2;      // 51.2 MB
  float*  candS = (float*) (ws + off); off += (size_t)BATCH * CAND_PER_ROW * 4;
  int*    candI = (int*)   (ws + off); off += (size_t)BATCH * CAND_PER_ROW * 4;

  float* outGather = (float*)d_out;                                   // [512,16,256]
  float* outIdx    = (float*)d_out + (size_t)BATCH * TOPK * DIM;      // [512,16]

  lrb_proj_norm_q<<<dim3(BATCH), dim3(256), 0, stream>>>(z_ego, W, qn);
  lrb_norm_bank<<<dim3(NBANK), dim3(256), 0, stream>>>(bank, bankn);
  lrb_gemm_score_topk<<<dim3(NCHUNK, BATCH / 16), dim3(256), 0, stream>>>(
      qn, bankn, simvt, candS, candI);
  lrb_finalize<<<dim3(BATCH), dim3(256), 0, stream>>>(candS, candI, bank,
                                                      outGather, outIdx);
}
